// CompositionalLoss_69140383531603
// MI455X (gfx1250) — matrix operand order repaired and checked by reading, not compiled, not run on hardware
//
#include <hip/hip_runtime.h>

typedef __attribute__((ext_vector_type(16))) _Float16 v16h;
typedef __attribute__((ext_vector_type(8)))  float    v8f;

#define NJ     21      // joints
#define NPAIR  210     // 21 choose 2
#define MT     14      // 224 padded pair-rows / 16
#define KPAD   40      // halves per Dh row (>=32, 80B stride -> 16B-aligned frags, 2-way banks)
#define WAVES  4
#define TILE_B 32      // batches per wave
#define BLK_B  (WAVES * TILE_B)

// ---------------------------------------------------------------------------
// Compile-time A-fragment table for v_wmma_f32_16x16x32_f16.
// A is the pair-difference matrix: row m = pair (u,v) (combinations order),
// +1 at column u, -1 at column v, zero elsewhere; rows >= 210 are zero pad.
// A 16x32 layout used here (contiguous k, same as B): lane L holds row M = L&15, element e holds
// K = (L>>4)*16 + e.  Values are f16 bit patterns.
// ---------------------------------------------------------------------------
struct AFrag { unsigned short v[MT][32][16]; };

constexpr AFrag makeAF() {
    AFrag t{};
    for (int mt = 0; mt < MT; ++mt) {
        for (int lane = 0; lane < 32; ++lane) {
            int m  = mt * 16 + (lane & 15);
            int hi = lane >> 4;
            int u = -1, vv = -1;
            if (m < NPAIR) {
                int r = m, uu = 0;
                while (r >= 20 - uu) { r -= 20 - uu; ++uu; }
                u = uu; vv = uu + 1 + r;
            }
            for (int e = 0; e < 16; ++e) {
                int k = hi * 16 + e;
                unsigned short val = 0;
                if (m < NPAIR) {
                    if (k == u)       val = 0x3C00;  // +1.0h
                    else if (k == vv) val = 0xBC00;  // -1.0h
                }
                t.v[mt][lane][e] = val;
            }
        }
    }
    return t;
}

__constant__ __attribute__((aligned(32))) AFrag AF = makeAF();

// ---------------------------------------------------------------------------
__global__ __launch_bounds__(128)
void comploss_kernel(const float* __restrict__ in, const float* __restrict__ tg,
                     float* __restrict__ out, int Btotal)
{
    __shared__ __attribute__((aligned(64))) float smem[WAVES][4032];

    const int tid  = threadIdx.x;
    const int w    = tid >> 5;
    const int lane = tid & 31;

    float*     xs = smem[w];             // 2016 f32: 32 batches x 63 bone floats
    float*     ts = smem[w] + 2016;      // 2016 f32: 32 batches x 63 target floats
    _Float16*  Dh = (_Float16*)smem[w];  // overlays xs after stage 1 (3*32*KPAD halves)

    const long b0    = ((long)blockIdx.x * WAVES + w) * TILE_B;
    const long limit = (long)Btotal * 63;

    // ---- stage 0: async global -> LDS copy (CDNA5 direct-to-LDS path) ----
    if (b0 + TILE_B <= (long)Btotal) {
        // Per-lane base addresses; IOFFSET is added to BOTH the global and
        // LDS addresses, so immediate offsets step through the 8064B block
        // in 512B chunks (32 lanes x 16B).  504 b128 lanes total: 15 full
        // chunks + one 24-lane chunk (EXEC-masked).
        const uint64_t gx = (uint64_t)(uintptr_t)(in + b0 * 63) + (uint32_t)(lane * 16);
        const uint64_t gt = (uint64_t)(uintptr_t)(tg + b0 * 63) + (uint32_t)(lane * 16);
        const uint32_t lx = (uint32_t)(uintptr_t)xs + (uint32_t)(lane * 16);
        const uint32_t lt = (uint32_t)(uintptr_t)ts + (uint32_t)(lane * 16);
#define A2(OFF)                                                                       \
    do {                                                                              \
        asm volatile("global_load_async_to_lds_b128 %0, %1, off offset:" #OFF        \
                     :: "v"(lx), "v"(gx) : "memory");                                 \
        asm volatile("global_load_async_to_lds_b128 %0, %1, off offset:" #OFF        \
                     :: "v"(lt), "v"(gt) : "memory");                                 \
    } while (0)
        A2(0);    A2(512);  A2(1024); A2(1536); A2(2048);
        A2(2560); A2(3072); A2(3584); A2(4096); A2(4608);
        A2(5120); A2(5632); A2(6144); A2(6656); A2(7168);
        if (lane < 24) { A2(7680); }
#undef A2
    } else {                                     // tail / OOB: zero-fill, sync loads
        for (int idx = lane; idx < 2016; idx += 32) {
            long g = b0 * 63 + idx;
            float xv = 0.f, tv = 0.f;
            if (g >= 0 && g < limit) { xv = in[g]; tv = tg[g]; }
            xs[idx] = xv; ts[idx] = tv;
        }
    }
    asm volatile("s_wait_asynccnt 0x0" ::: "memory");
    __syncthreads();

    // ---- stage 1: per-lane (lane = batch) tree prefix ---------------------
    float X[63];
    #pragma unroll
    for (int i = 0; i < 63; ++i) X[i] = xs[lane * 63 + i];
    __syncthreads();   // everyone done reading xs before Dh overlays it

    // zero Dh including K padding (3*32*KPAD halves = 1920 dwords)
    {
        unsigned int* dz = (unsigned int*)Dh;
        #pragma unroll
        for (int it = 0; it < 60; ++it) dz[it * 32 + lane] = 0u;
    }

    // A[j] = x[j] + A[parent(j)], topological order (root Hip=3 first, no-op)
    static constexpr int TOPO[20] = {0,1,9,18,8,12,17,7,16,2,11,20,5,14,4,13,10,19,6,15};
    static constexpr int PAR[NJ]  = {3,0,12,3,11,7,4,9,1,3,5,8,1,20,16,13,18,1,3,14,17};
    #pragma unroll
    for (int s = 0; s < 20; ++s) {
        int c = TOPO[s], p = PAR[c];
        X[c*3+0] += X[p*3+0];
        X[c*3+1] += X[p*3+1];
        X[c*3+2] += X[p*3+2];
    }

    // D[j] = A[j] - t[j]  ->  f16 LDS tile Dh[d][batch][k]
    #pragma unroll
    for (int j = 0; j < NJ; ++j) {
        #pragma unroll
        for (int d = 0; d < 3; ++d) {
            Dh[(d * 32 + lane) * KPAD + j] =
                (_Float16)(X[j*3 + d] - ts[lane * 63 + j*3 + d]);
        }
    }
    __syncthreads();

    // ---- stage 2: WMMA pair differences + L1 accumulate -------------------
    // B-fragment (contiguous k, same order as A): lane L element e holds K = (L>>4)*16+e,
    // N = L&15  -> contiguous 16 halves per lane in our [d][n][k] layout.
    // C/D: lane L vgpr r holds M = r + (L>>4)*8, N = L&15.
    const int n16 = lane & 15, hi = lane >> 4;
    float acc0 = 0.f, acc1 = 0.f, acc2 = 0.f, acc3 = 0.f;  // break add chains
    #pragma unroll
    for (int mt = 0; mt < MT; ++mt) {
        const v16h a = *(const v16h*)(&AF.v[mt][lane][0]);
        #pragma unroll
        for (int d = 0; d < 3; ++d) {
            #pragma unroll
            for (int nt = 0; nt < 2; ++nt) {
                const v16h b =
                    *(const v16h*)(Dh + ((d * 32 + nt * 16 + n16) * KPAD + hi * 16));
                v8f c = {};
                c = __builtin_amdgcn_wmma_f32_16x16x32_f16(
                        /*neg_a=*/false, a, /*neg_b=*/false, b,
                        /*c_mod=*/(short)0, c, /*reuse_a=*/false, /*reuse_b=*/false);
                acc0 += __builtin_fabsf(c[0]);
                acc1 += __builtin_fabsf(c[1]);
                acc2 += __builtin_fabsf(c[2]);
                acc3 += __builtin_fabsf(c[3]);
                acc0 += __builtin_fabsf(c[4]);
                acc1 += __builtin_fabsf(c[5]);
                acc2 += __builtin_fabsf(c[6]);
                acc3 += __builtin_fabsf(c[7]);
            }
        }
    }
    float acc = (acc0 + acc1) + (acc2 + acc3);

    // ---- wave reduce + global accumulate ----------------------------------
    #pragma unroll
    for (int off = 16; off > 0; off >>= 1)
        acc += __shfl_down(acc, off, 32);
    if (lane == 0)
        atomicAdd(out, acc / (float)Btotal);
}

// ---------------------------------------------------------------------------
extern "C" void kernel_launch(void* const* d_in, const int* in_sizes, int n_in,
                              void* d_out, int out_size, void* d_ws, size_t ws_size,
                              hipStream_t stream)
{
    const float* in = (const float*)d_in[0];
    const float* tg = (const float*)d_in[1];
    float* out = (float*)d_out;

    const int Btotal = in_sizes[0] / 63;   // 21 joints * 3 dims
    hipMemsetAsync(d_out, 0, sizeof(float), stream);

    if (Btotal > 0) {
        const int nblocks = (Btotal + BLK_B - 1) / BLK_B;
        comploss_kernel<<<nblocks, 128, 0, stream>>>(in, tg, out, Btotal);
    }
}
